// SelfAtten_11836929868166
// MI455X (gfx1250) — matrix operand order repaired and checked by reading, not compiled, not run on hardware
//
#include <hip/hip_runtime.h>
#include <hip/hip_bf16.h>

// ---------------- problem constants ----------------
#define B_   4
#define C_   384
#define H_   128
#define W_   128
#define L_   (H_ * W_)     // 16384
#define NH_  8
#define D_   48            // C_/NH_

typedef __attribute__((ext_vector_type(16))) __bf16 v16bf;
typedef __attribute__((ext_vector_type(8)))  __bf16 bf16x8;
typedef __attribute__((ext_vector_type(8)))  float  v8f;

static __device__ __forceinline__ v8f wmma_bf16(v16bf a, v16bf b, v8f c) {
    return __builtin_amdgcn_wmma_f32_16x16x32_bf16(
        /*neg_a=*/false, a, /*neg_b=*/false, b,
        /*c_mod=*/(short)0, c, /*reuse_a=*/false, /*reuse_b=*/false);
}

static __device__ __forceinline__ v16bf make_frag(bf16x8 lo, bf16x8 hi) {
    v16bf r;
#pragma unroll
    for (int i = 0; i < 8; ++i) { r[i] = lo[i]; r[8 + i] = hi[i]; }
    return r;
}

// ============================================================
// K0: convert wq/wk/wv (f32) -> bf16, packed [3][C][C]
// ============================================================
__global__ void k_wconvert(const float* __restrict__ wq,
                           const float* __restrict__ wk,
                           const float* __restrict__ wv,
                           __bf16* __restrict__ o) {
    int i = blockIdx.x * 256 + threadIdx.x;
    const int nn = C_ * C_;
    if (i < nn)            o[i] = (__bf16)wq[i];
    else if (i < 2 * nn)   o[i] = (__bf16)wk[i - nn];
    else if (i < 3 * nn)   o[i] = (__bf16)wv[i - 2 * nn];
}

// ============================================================
// K1: LayerNorm over channel dim, f32 in -> bf16 out, [B][C][L]
// ============================================================
__global__ __launch_bounds__(256)
void k_ln_f32_to_bf16(const float* __restrict__ x,
                      const float* __restrict__ w,
                      const float* __restrict__ bias,
                      __bf16* __restrict__ out) {
    int p = blockIdx.x * 256 + threadIdx.x;      // over B*L
    if (p >= B_ * L_) return;
    int b = p / L_, l = p % L_;
    const float* xb = x + (size_t)b * C_ * L_ + l;
    float s = 0.f, s2 = 0.f;
    for (int c = 0; c < C_; ++c) {
        float v = xb[(size_t)c * L_];
        s += v; s2 += v * v;
    }
    float mu = s * (1.0f / C_);
    float var = fmaxf(s2 * (1.0f / C_) - mu * mu, 0.f);
    float rstd = rsqrtf(var + 1e-5f);
    __bf16* ob = out + (size_t)b * C_ * L_ + l;
    for (int c = 0; c < C_; ++c) {
        float v = (xb[(size_t)c * L_] - mu) * rstd * w[c] + bias[c];
        ob[(size_t)c * L_] = (__bf16)v;
    }
}

// ============================================================
// K2: QKV 1x1-conv GEMM, bf16 WMMA, f32 accum.
// out[o][l] = sum_c W[o][c] * xn[c][l]   per (batch, q/k/v)
// grid: (L/64, 3*B); block 256 = 8 waves.
// Block computes ALL 384 output channels for one 64-pixel tile:
// wave w -> o rows [w*48, w*48+48)  = 3 (o) x 4 (l) = 12 acc tiles.
// Entire 64l x 384c xn tile staged to LDS ONCE (50 KB), single
// barrier pair; K-loop runs from LDS.  Epilogue transposes through
// the (dead) LDS tile for coalesced b128 global stores.
// ============================================================
#define SXSTRIDE 392            // padded row stride (bf16 units), even
#define SX(l, c) sx[(l) * SXSTRIDE + (c)]

__global__ __launch_bounds__(256)
void k_qkv_gemm(const __bf16* __restrict__ xn,
                const __bf16* __restrict__ wbf,   // [3][C][C]
                __bf16* __restrict__ qpre,
                __bf16* __restrict__ kpre,
                __bf16* __restrict__ vpre) {
    const int lt    = blockIdx.x * 64;
    const int which = blockIdx.y / B_;
    const int b     = blockIdx.y % B_;

    const __bf16* Wm = wbf + (size_t)which * C_ * C_;
    __bf16* O = (which == 0) ? qpre : (which == 1) ? kpre : vpre;

    const int tid  = threadIdx.x;
    const int wave = tid >> 5;
    const int lane = tid & 31;

    __shared__ __bf16 sx[64 * SXSTRIDE];    // 50176 B

    const __bf16* xb = xn + (size_t)b * C_ * L_;

    // ---- stage full 64l x 384c tile, transposed, paired-channel b32 ----
#pragma unroll
    for (int j = 0; j < 6; ++j) {
        int tk    = tid + 256 * j;           // 0..1535
        int cpair = tk >> 3;                 // 0..191
        int lc    = (tk & 7) * 8;            // 0..56
        int c     = cpair * 2;
        const __bf16* s0 = xb + (size_t)c * L_ + lt + lc;
        bf16x8 r0 = *(const bf16x8*)s0;
        bf16x8 r1 = *(const bf16x8*)(s0 + L_);
#pragma unroll
        for (int i = 0; i < 8; ++i) {
            union { __bf16 h[2]; unsigned u; } p;
            p.h[0] = r0[i]; p.h[1] = r1[i];
            *(unsigned*)&SX(lc + i, c) = p.u;
        }
    }
    __syncthreads();

    const int m     = lane & 15;
    const int kbase = (lane >> 4) ? 8 : 0;
    const int n     = lane & 15;
    const int kh    = (lane >> 4) * 16;
    const int ow0   = wave * 48;             // this wave's o base

    v8f acc[3][4] = {};                      // [o-subtile][l-subtile]

    for (int kk = 0; kk < C_; kk += 32) {
        v16bf a[3];
#pragma unroll
        for (int i = 0; i < 3; ++i) {
            const __bf16* wrow = Wm + (size_t)(ow0 + i * 16 + m) * C_ + kk;
            a[i] = make_frag(*(const bf16x8*)(wrow + kbase),
                             *(const bf16x8*)(wrow + kbase + 16));
        }
#pragma unroll
        for (int t = 0; t < 4; ++t) {
            const __bf16* brow = &SX(t * 16 + n, kk + kbase);
            v16bf bb = make_frag(*(const bf16x8*)(brow),
                                 *(const bf16x8*)(brow + 16));
#pragma unroll
            for (int i = 0; i < 3; ++i)
                acc[i][t] = wmma_bf16(a[i], bb, acc[i][t]);
        }
    }

    // ---- epilogue: through LDS (region is dead now) for coalesced stores
    __syncthreads();
    {
        __bf16* so = sx;                     // view: so[o*64 + l], 384*64 elems
        const int mo = (lane >> 4) * 8;
#pragma unroll
        for (int i = 0; i < 3; ++i)
#pragma unroll
            for (int t = 0; t < 4; ++t)
#pragma unroll
                for (int r = 0; r < 8; ++r)
                    so[(ow0 + i * 16 + mo + r) * 64 + t * 16 + n] =
                        (__bf16)acc[i][t][r];
    }
    __syncthreads();
#pragma unroll
    for (int j = 0; j < 12; ++j) {
        int ch  = tid + 256 * j;             // 0..3071
        int o   = ch >> 3;                   // 0..383
        int seg = (ch & 7) * 8;              // 0..56
        bf16x8 vdat = *(const bf16x8*)&sx[o * 64 + seg];
        *(bf16x8*)&O[((size_t)b * C_ + o) * L_ + lt + seg] = vdat;
    }
}

// ============================================================
// K3: depthwise 3x3, pad=1, bf16 -> bf16, f32 weights
// ============================================================
__global__ __launch_bounds__(256)
void k_dwconv(const __bf16* __restrict__ in,
              const float* __restrict__ wt,     // [C][9]
              __bf16* __restrict__ out) {
    size_t idx = (size_t)blockIdx.x * 256 + threadIdx.x;   // over B*C*L
    if (idx >= (size_t)B_ * C_ * L_) return;
    int l = (int)(idx % L_);
    size_t bc = idx / L_;
    int c = (int)(bc % C_);
    int h = l >> 7, w0 = l & (W_ - 1);
    const __bf16* base = in + bc * L_;
    if (h + 2 < H_) __builtin_prefetch(base + (h + 2) * W_ + w0, 0, 1);
    const float* k9 = wt + c * 9;
    float s = 0.f;
#pragma unroll
    for (int dy = -1; dy <= 1; ++dy) {
        int y = h + dy; if (y < 0 || y >= H_) continue;
#pragma unroll
        for (int dx = -1; dx <= 1; ++dx) {
            int x = w0 + dx; if (x < 0 || x >= W_) continue;
            s += (float)base[y * W_ + x] * k9[(dy + 1) * 3 + (dx + 1)];
        }
    }
    out[idx] = (__bf16)s;
}

// ============================================================
// K4: per-row (b,c) L2 normalize over L, in place; optional *scale[head]
// ============================================================
__global__ __launch_bounds__(256)
void k_rownorm(__bf16* __restrict__ data, const float* __restrict__ scale) {
    int row = blockIdx.x;                       // B*C rows
    int c = row % C_;
    int head = c / D_;
    __bf16* p = data + (size_t)row * L_;
    float ss = 0.f;
    for (int i = threadIdx.x; i < L_; i += 256) {
        float v = (float)p[i]; ss += v * v;
    }
    __shared__ float r[256];
    r[threadIdx.x] = ss; __syncthreads();
    for (int s2 = 128; s2 > 0; s2 >>= 1) {
        if (threadIdx.x < s2) r[threadIdx.x] += r[threadIdx.x + s2];
        __syncthreads();
    }
    float f = 1.0f / fmaxf(sqrtf(r[0]), 1e-12f);
    if (scale) f *= scale[head];
    for (int i = threadIdx.x; i < L_; i += 256)
        p[i] = (__bf16)((float)p[i] * f);
}

// ============================================================
// K5a: attn = softmax(qn . kn^T) per (b,head).  Contraction over L
// is contiguous for BOTH fragments.  8 waves split L, 9 acc tiles
// each, cross-wave reduce with LDS atomicAdd (ds_add_f32), softmax,
// store bf16 attn [48][64] (K-padded with zeros).
// ============================================================
__global__ __launch_bounds__(256)
void k_attn_qk(const __bf16* __restrict__ qn,
               const __bf16* __restrict__ kn,
               __bf16* __restrict__ attn) {
    const int bh = blockIdx.x;                 // B*NH blocks
    const int b = bh / NH_, h = bh % NH_;
    const __bf16* qb = qn + ((size_t)b * C_ + h * D_) * L_;
    const __bf16* kb = kn + ((size_t)b * C_ + h * D_) * L_;

    const int tid = threadIdx.x, wave = tid >> 5, lane = tid & 31;
    const int m = lane & 15, kbase = (lane >> 4) ? 8 : 0;
    const int n = lane & 15, kh = (lane >> 4) * 16;

    __shared__ float red[D_ * D_];
    for (int i = tid; i < D_ * D_; i += 256) red[i] = 0.f;
    __syncthreads();

    v8f acc[3][3] = {};
    const int lbeg = wave * (L_ / 8);
    for (int l = lbeg; l < lbeg + L_ / 8; l += 32) {
        v16bf afr[3], bfr[3];
#pragma unroll
        for (int i = 0; i < 3; ++i) {
            const __bf16* ar = qb + (size_t)(i * 16 + m) * L_ + l;
            afr[i] = make_frag(*(const bf16x8*)(ar + kbase),
                               *(const bf16x8*)(ar + kbase + 16));
            const __bf16* br = kb + (size_t)(i * 16 + n) * L_ + l + kbase;
            bfr[i] = make_frag(*(const bf16x8*)(br),
                               *(const bf16x8*)(br + 16));
        }
#pragma unroll
        for (int i = 0; i < 3; ++i)
#pragma unroll
            for (int j = 0; j < 3; ++j)
                acc[i][j] = wmma_bf16(afr[i], bfr[j], acc[i][j]);
    }

    const int mo = (lane >> 4) * 8;
#pragma unroll
    for (int i = 0; i < 3; ++i)
#pragma unroll
        for (int j = 0; j < 3; ++j)
#pragma unroll
            for (int r = 0; r < 8; ++r) {
                int d = i * 16 + mo + r;
                int e = j * 16 + n;
                atomicAdd(&red[d * D_ + e], acc[i][j][r]);
            }
    __syncthreads();

    // softmax rows, write bf16 [48][64] zero-padded
    if (tid < D_) {
        float mx = -1e30f;
        for (int e = 0; e < D_; ++e) mx = fmaxf(mx, red[tid * D_ + e]);
        float ex[D_], sum = 0.f;
        for (int e = 0; e < D_; ++e) {
            ex[e] = __expf(red[tid * D_ + e] - mx); sum += ex[e];
        }
        float inv = 1.0f / sum;
        __bf16* arow = attn + (size_t)bh * D_ * 64 + tid * 64;
        for (int e = 0; e < D_; ++e) arow[e] = (__bf16)(ex[e] * inv);
        for (int e = D_; e < 64; ++e) arow[e] = (__bf16)0.f;
    }
}

// ============================================================
// K5b: out[d][l] = sum_e attn[d][e] * v[e][l], f32 output
// grid (L/64, B*NH); 4 waves, each a 16-l column slice, 3 d tiles.
// ============================================================
__global__ __launch_bounds__(128)
void k_attn_v(const __bf16* __restrict__ attn,   // [BH][48][64]
              const __bf16* __restrict__ v,
              float* __restrict__ out) {
    const int bh = blockIdx.y;
    const int b = bh / NH_, h = bh % NH_;
    const int lt = blockIdx.x * 64;
    const int tid = threadIdx.x, wave = tid >> 5, lane = tid & 31;

    const __bf16* A  = attn + (size_t)bh * D_ * 64;
    const __bf16* vb = v + ((size_t)b * C_ + h * D_) * L_;

    const int m = lane & 15, kbase = (lane >> 4) ? 8 : 0;
    const int n = lane & 15, kh = (lane >> 4) * 16;
    const int lcol = lt + wave * 16 + n;

    v8f acc[3] = {};
#pragma unroll
    for (int kk = 0; kk < 64; kk += 32) {
        // B fragment: column lcol, K = e (strided by L; clamp keeps it
        // in-bounds; padded attn zeros kill the garbage contribution)
        v16bf bfr;
#pragma unroll
        for (int i = 0; i < 16; ++i) {
            int e = kk + kbase + (i & 7) + ((i >> 3) << 4); e = (e < D_) ? e : (D_ - 1);
            bfr[i] = vb[(size_t)e * L_ + lcol];
        }
#pragma unroll
        for (int i = 0; i < 3; ++i) {
            const __bf16* ar = A + (size_t)(i * 16 + m) * 64 + kk;
            v16bf afr = make_frag(*(const bf16x8*)(ar + kbase),
                                  *(const bf16x8*)(ar + kbase + 16));
            acc[i] = wmma_bf16(afr, bfr, acc[i]);
        }
    }
    const int mo = (lane >> 4) * 8;
#pragma unroll
    for (int i = 0; i < 3; ++i)
#pragma unroll
        for (int r = 0; r < 8; ++r) {
            int d = i * 16 + mo + r;
            out[((size_t)b * C_ + h * D_ + d) * L_ + lcol] = acc[i][r];
        }
}

// ============================================================
// K6: final LayerNorm, f32 -> f32
// ============================================================
__global__ __launch_bounds__(256)
void k_ln_f32_to_f32(const float* __restrict__ x,
                     const float* __restrict__ w,
                     const float* __restrict__ bias,
                     float* __restrict__ out) {
    int p = blockIdx.x * 256 + threadIdx.x;
    if (p >= B_ * L_) return;
    int b = p / L_, l = p % L_;
    const float* xb = x + (size_t)b * C_ * L_ + l;
    float s = 0.f, s2 = 0.f;
    for (int c = 0; c < C_; ++c) {
        float v = xb[(size_t)c * L_];
        s += v; s2 += v * v;
    }
    float mu = s * (1.0f / C_);
    float var = fmaxf(s2 * (1.0f / C_) - mu * mu, 0.f);
    float rstd = rsqrtf(var + 1e-5f);
    float* ob = out + (size_t)b * C_ * L_ + l;
    for (int c = 0; c < C_; ++c)
        ob[(size_t)c * L_] = (xb[(size_t)c * L_] - mu) * rstd * w[c] + bias[c];
}

// ============================================================
extern "C" void kernel_launch(void* const* d_in, const int* in_sizes, int n_in,
                              void* d_out, int out_size, void* d_ws, size_t ws_size,
                              hipStream_t stream) {
    const float* x     = (const float*)d_in[0];
    const float* wq    = (const float*)d_in[1];
    const float* wk    = (const float*)d_in[2];
    const float* wv    = (const float*)d_in[3];
    const float* dwq   = (const float*)d_in[4];
    const float* dwk   = (const float*)d_in[5];
    const float* dwv   = (const float*)d_in[6];
    const float* scale = (const float*)d_in[7];
    const float* ln1w  = (const float*)d_in[8];
    const float* ln1b  = (const float*)d_in[9];
    const float* ln2w  = (const float*)d_in[10];
    const float* ln2b  = (const float*)d_in[11];
    float* out = (float*)d_out;

    // workspace layout (bf16 plane = B*C*L*2 bytes = 50.33 MB)
    const size_t S1 = (size_t)B_ * C_ * L_ * sizeof(__bf16);
    char* ws = (char*)d_ws;
    __bf16* xn   = (__bf16*)(ws);
    __bf16* qpre = (__bf16*)(ws + 1 * S1);
    __bf16* kpre = (__bf16*)(ws + 2 * S1);
    __bf16* vpre = (__bf16*)(ws + 3 * S1);
    __bf16* qdw  = (__bf16*)(ws + 4 * S1);
    __bf16* kdw  = (__bf16*)(ws + 5 * S1);
    __bf16* vdw  = (__bf16*)(ws + 6 * S1);
    __bf16* attn = (__bf16*)(ws + 7 * S1);                 // 32*48*64 bf16
    __bf16* wbf  = (__bf16*)(ws + 7 * S1 + (1 << 20));     // [3][C][C] bf16
    float*  aout = (float*)(ws);   // reuses xn+qpre (dead after K2/K3): B*C*L f32 == 2*S1

    // K0: weights -> bf16
    k_wconvert<<<(3 * C_ * C_ + 255) / 256, 256, 0, stream>>>(wq, wk, wv, wbf);
    // K1: LN1 -> bf16
    k_ln_f32_to_bf16<<<(B_ * L_ + 255) / 256, 256, 0, stream>>>(x, ln1w, ln1b, xn);
    // K2: QKV GEMMs (block covers all 384 out-channels of a 64-pixel tile)
    {
        dim3 g(L_ / 64, 3 * B_);
        k_qkv_gemm<<<g, 256, 0, stream>>>(xn, wbf, qpre, kpre, vpre);
    }
    // K3: depthwise 3x3 (q, k, v)
    {
        int nb = (int)(((size_t)B_ * C_ * L_ + 255) / 256);
        k_dwconv<<<nb, 256, 0, stream>>>(qpre, dwq, qdw);
        k_dwconv<<<nb, 256, 0, stream>>>(kpre, dwk, kdw);
        k_dwconv<<<nb, 256, 0, stream>>>(vpre, dwv, vdw);
    }
    // K4: L2 row normalize (scale folded into q)
    k_rownorm<<<B_ * C_, 256, 0, stream>>>(qdw, scale);
    k_rownorm<<<B_ * C_, 256, 0, stream>>>(kdw, nullptr);
    // K5a: attn + softmax
    k_attn_qk<<<B_ * NH_, 256, 0, stream>>>(qdw, kdw, attn);
    // K5b: attn @ v
    {
        dim3 g(L_ / 64, B_ * NH_);
        k_attn_v<<<g, 128, 0, stream>>>(attn, vdw, aout);
    }
    // K6: LN2 -> output
    k_ln_f32_to_f32<<<(B_ * L_ + 255) / 256, 256, 0, stream>>>(aout, ln2w, ln2b, out);
}
